// GraphTransformerLayer_57896159150476
// MI455X (gfx1250) — matrix operand order repaired and checked by reading, not compiled, not run on hardware
//
#include <hip/hip_runtime.h>

#define EMBED 256
#define HEADS 8
#define HDIM  32
#define BATCH 8
#define SEQ   1024

typedef _Float16 v16h __attribute__((ext_vector_type(16)));
typedef _Float16 v8h  __attribute__((ext_vector_type(8)));
typedef float    v8f  __attribute__((ext_vector_type(8)));

// ---- WMMA operand loaders (CDNA5 16x16x32 f16 layouts, wave32) ----
// A matrix 16x32 (M x K), row-major source with stride `strideH` halves:
//   lanes 0-15 : V0-3 = K0..7  (bytes 0..15),  V4-7 = K16..23 (bytes 32..47)
//   lanes16-31 : V0-3 = K8..15 (bytes 16..31), V4-7 = K24..31 (bytes 48..63)
__device__ __forceinline__ v16h load_a16x32(const _Float16* base, int strideH, int lane) {
  int row = lane & 15;
  int sel = (lane >> 4) & 1;
  const _Float16* p = base + row * strideH + sel * 8;
  v8h lo = *(const v8h*)(p);
  v8h hi = *(const v8h*)(p + 16);
  v16h r;
#pragma unroll
  for (int i = 0; i < 8; ++i) { r[i] = lo[i]; r[i + 8] = hi[i]; }
  return r;
}
// B matrix 32x16 (K x N), source stored row-major as [N][K] with stride `strideH`:
//   lanes 0-15 : V0-3 = K0..7, V4-7 = K16..23 ; lanes 16-31 : V0-3 = K8..15, V4-7 = K24..31
__device__ __forceinline__ v16h load_b32x16(const _Float16* base, int strideH, int lane) {
  int row = lane & 15;
  int sel = (lane >> 4) & 1;
  const _Float16* p = base + row * strideH + sel * 8;
  v8h lo = *(const v8h*)(p);
  v8h hi = *(const v8h*)(p + 16);
  v16h r;
#pragma unroll
  for (int i = 0; i < 8; ++i) { r[i] = lo[i]; r[i + 8] = hi[i]; }
  return r;
}

__device__ __forceinline__ v8f wmma16(v16h a, v16h b, v8f c) {
  return __builtin_amdgcn_wmma_f32_16x16x32_f16(false, a, false, b, (short)0, c, false, false);
}

__device__ __forceinline__ float redmax16(float v) {
#pragma unroll
  for (int m = 1; m < 16; m <<= 1) v = fmaxf(v, __shfl_xor(v, m, 32));
  return v;
}
__device__ __forceinline__ float redsum16(float v) {
#pragma unroll
  for (int m = 1; m < 16; m <<= 1) v += __shfl_xor(v, m, 32);
  return v;
}

// Low 32 bits of a flat pointer into the LDS aperture are the workgroup-relative
// LDS byte offset (ISA 10.2).
__device__ __forceinline__ unsigned lds_addr32(const void* p) {
  return (unsigned)(unsigned long long)p;
}
// gfx1250 async global->LDS copy (ASYNCcnt-tracked, bypasses VGPRs).
__device__ __forceinline__ void async_g2l_b128(unsigned lds_off, const void* gaddr) {
  asm volatile("global_load_async_to_lds_b128 %0, %1, off"
               :: "v"(lds_off), "v"(gaddr) : "memory");
}

// ---- stage 0: f32 -> f16 conversions ----
__global__ void gt_f32_to_f16(const float* __restrict__ src, _Float16* __restrict__ dst, int n) {
  int i = blockIdx.x * blockDim.x + threadIdx.x;
  if (i < n) dst[i] = (_Float16)src[i];
}
// Wt[n][k] = W[k][n], 256x256, to f16 (makes GEMM B-operand loads contiguous)
__global__ void gt_transpose_w(const float* __restrict__ W, _Float16* __restrict__ Wt) {
  int i = blockIdx.x * blockDim.x + threadIdx.x;   // 65536 threads
  int n = i >> 8, k = i & 255;
  Wt[n * 256 + k] = (_Float16)W[k * 256 + n];
}

// ---- stage 1: QKV projection GEMM (one wave per 16x16 output tile) ----
// Q,K written as [B][H][N][D] f16; V written transposed [B][H][D][N] f16.
__global__ __launch_bounds__(32)
void gt_qkv(const _Float16* __restrict__ xh,
            const _Float16* __restrict__ Wtq, const _Float16* __restrict__ Wtk,
            const _Float16* __restrict__ Wtv,
            const float* __restrict__ bq, const float* __restrict__ bk,
            const float* __restrict__ bv,
            _Float16* __restrict__ Q, _Float16* __restrict__ K, _Float16* __restrict__ Vt) {
  int lane = threadIdx.x;
  int col0 = blockIdx.x * 16;
  int row0 = blockIdx.y * 16;
  int which = blockIdx.z;
  const _Float16* Wt = (which == 0) ? Wtq : (which == 1) ? Wtk : Wtv;
  const float*   bia = (which == 0) ? bq  : (which == 1) ? bk  : bv;
  v8f c = {};
#pragma unroll
  for (int kk = 0; kk < EMBED; kk += 32) {
    v16h a = load_a16x32(xh + row0 * EMBED + kk, EMBED, lane);
    v16h b = load_b32x16(Wt + col0 * EMBED + kk, EMBED, lane);
    c = wmma16(a, b, c);
  }
  int j = lane & 15, g = lane >> 4;
  float bval = bia[col0 + j];
  int h = col0 >> 5;
  int d = (col0 & 31) + j;
#pragma unroll
  for (int r = 0; r < 8; ++r) {
    int rn = row0 + r + 8 * g;
    int b_ = rn >> 10, n = rn & 1023;
    _Float16 hv = (_Float16)(c[r] + bval);
    if (which == 0)      Q [(((b_ * HEADS + h) * SEQ + n) * HDIM) + d] = hv;
    else if (which == 1) K [(((b_ * HEADS + h) * SEQ + n) * HDIM) + d] = hv;
    else                 Vt[(((b_ * HEADS + h) * HDIM + d) * SEQ) + n] = hv;
  }
}

// ---- stage 2: fused flash attention + adjacency bias ----
// One workgroup = (batch b, 16 query rows), 8 waves = 8 heads.
// adj tile streamed ONCE via double-buffered async global->LDS copies.
__global__ __launch_bounds__(256)
void gt_attn(const _Float16* __restrict__ Q, const _Float16* __restrict__ K,
             const _Float16* __restrict__ Vt,
             const float* __restrict__ adj, const float* __restrict__ Wa,
             const float* __restrict__ ba, _Float16* __restrict__ ctx) {
  __shared__ __align__(16) float4   adjT[2][16][32];       // 16 KB, double-buffered
  __shared__ __align__(16) _Float16 pbuf[HEADS][16][32];   // 8 KB
  int tid = threadIdx.x;
  int h = tid >> 5, lane = tid & 31;
  int bb = blockIdx.x >> 6;            // SEQ/16 = 64 row tiles per batch
  int n0 = (blockIdx.x & 63) << 4;
  const float scale = 0.17677669529663688f;  // 1/sqrt(32)
  float wa0 = Wa[0 * HEADS + h], wa1 = Wa[1 * HEADS + h];
  float wa2 = Wa[2 * HEADS + h], wa3 = Wa[3 * HEADS + h];
  float bah = ba[h];

  const float4* adj4 = (const float4*)adj;
  // this thread's two (row, col) slots of the 16x32 adj tile
  int ir0 = tid >> 5,          jc0 = tid & 31;        // e = tid
  int ir1 = (tid + 256) >> 5,  jc1 = tid & 31;        // e = tid + 256
  size_t arow0 = (size_t)(bb * SEQ + n0 + ir0) * SEQ;
  size_t arow1 = (size_t)(bb * SEQ + n0 + ir1) * SEQ;

  const _Float16* Qb = Q  + ((bb * HEADS + h) * SEQ + n0) * HDIM;
  const _Float16* Kb = K  + ((size_t)(bb * HEADS + h) * SEQ) * HDIM;
  const _Float16* Vb = Vt + ((size_t)(bb * HEADS + h) * HDIM) * SEQ;

  v16h qa = load_a16x32(Qb, HDIM, lane);
  v8f acc0 = {}, acc1 = {};
  float rmax[8], rsum[8];
#pragma unroll
  for (int r = 0; r < 8; ++r) { rmax[r] = -1e30f; rsum[r] = 0.f; }
  int j = lane & 15, g = lane >> 4;

  // prefetch tile 0 into buffer 0
  async_g2l_b128(lds_addr32(&adjT[0][ir0][jc0]), adj4 + arow0 + jc0);
  async_g2l_b128(lds_addr32(&adjT[0][ir1][jc1]), adj4 + arow1 + jc1);

  for (int it = 0; it < SEQ / 32; ++it) {
    int m0 = it * 32;
    int cur = it & 1;
    if (it + 1 < SEQ / 32) {
      // prefetch next tile into the other buffer (freed by last iteration's
      // trailing barrier), then wait for the CURRENT tile's 2 copies
      // (per-wave async loads complete in order => <=2 outstanding = cur done).
      int nm0 = m0 + 32;
      async_g2l_b128(lds_addr32(&adjT[cur ^ 1][ir0][jc0]), adj4 + arow0 + nm0 + jc0);
      async_g2l_b128(lds_addr32(&adjT[cur ^ 1][ir1][jc1]), adj4 + arow1 + nm0 + jc1);
      asm volatile("s_wait_asynccnt 0x2" ::: "memory");
    } else {
      asm volatile("s_wait_asynccnt 0x0" ::: "memory");
    }
    __syncthreads();   // all waves' portions of the current tile have landed

    v16h kb0 = load_b32x16(Kb + (size_t)m0 * HDIM, HDIM, lane);
    v16h kb1 = load_b32x16(Kb + (size_t)(m0 + 16) * HDIM, HDIM, lane);
    v8f z = {};
    v8f s0 = wmma16(qa, kb0, z);
    v8f s1 = wmma16(qa, kb1, z);

#pragma unroll
    for (int r = 0; r < 8; ++r) {
      int M = r + 8 * g;
      float4 a0 = adjT[cur][M][j];
      float4 a1 = adjT[cur][M][j + 16];
      float v0 = s0[r] * scale + a0.x * wa0 + a0.y * wa1 + a0.z * wa2 + a0.w * wa3 + bah;
      float v1 = s1[r] * scale + a1.x * wa0 + a1.y * wa1 + a1.z * wa2 + a1.w * wa3 + bah;
      float mx = redmax16(fmaxf(v0, v1));
      float nm = fmaxf(rmax[r], mx);
      float fac = __expf(rmax[r] - nm);
      rmax[r] = nm;
      float e0 = __expf(v0 - nm);
      float e1 = __expf(v1 - nm);
      rsum[r] = rsum[r] * fac + redsum16(e0 + e1);
      acc0[r] *= fac;
      acc1[r] *= fac;
      pbuf[h][M][j]      = (_Float16)e0;   // re-layout P through per-wave LDS
      pbuf[h][M][j + 16] = (_Float16)e1;
    }
    v16h pa  = load_a16x32(&pbuf[h][0][0], 32, lane);          // P as A operand
    v16h vb0 = load_b32x16(Vb + m0, SEQ, lane);                // V^T rows d=0..15
    v16h vb1 = load_b32x16(Vb + 16 * SEQ + m0, SEQ, lane);     // V^T rows d=16..31
    acc0 = wmma16(pa, vb0, acc0);
    acc1 = wmma16(pa, vb1, acc1);

    __syncthreads();   // everyone done reading adjT[cur] before it is re-filled
  }

#pragma unroll
  for (int r = 0; r < 8; ++r) {
    int M = r + 8 * g;
    float inv = 1.f / rsum[r];
    size_t rowidx = (size_t)(bb * SEQ + n0 + M) * EMBED + h * HDIM;
    ctx[rowidx + j]      = (_Float16)(acc0[r] * inv);
    ctx[rowidx + 16 + j] = (_Float16)(acc1[r] * inv);
  }
}

// ---- stage 3: output projection + residual ----
__global__ __launch_bounds__(32)
void gt_out(const _Float16* __restrict__ ctx, const _Float16* __restrict__ Wto,
            const float* __restrict__ bo, const float* __restrict__ x,
            float* __restrict__ out) {
  int lane = threadIdx.x;
  int col0 = blockIdx.x * 16;
  int row0 = blockIdx.y * 16;
  v8f c = {};
#pragma unroll
  for (int kk = 0; kk < EMBED; kk += 32) {
    v16h a = load_a16x32(ctx + row0 * EMBED + kk, EMBED, lane);
    v16h b = load_b32x16(Wto + col0 * EMBED + kk, EMBED, lane);
    c = wmma16(a, b, c);
  }
  int j = lane & 15, g = lane >> 4;
  float bval = bo[col0 + j];
#pragma unroll
  for (int r = 0; r < 8; ++r) {
    int rn = row0 + r + 8 * g;
    int idx = rn * EMBED + col0 + j;
    out[idx] = c[r] + bval + x[idx];
  }
}

extern "C" void kernel_launch(void* const* d_in, const int* in_sizes, int n_in,
                              void* d_out, int out_size, void* d_ws, size_t ws_size,
                              hipStream_t stream) {
  const float* x   = (const float*)d_in[0];
  const float* adj = (const float*)d_in[1];
  const float* Wq  = (const float*)d_in[2];
  const float* bq  = (const float*)d_in[3];
  const float* Wk  = (const float*)d_in[4];
  const float* bk  = (const float*)d_in[5];
  const float* Wv  = (const float*)d_in[6];
  const float* bv  = (const float*)d_in[7];
  const float* Wo  = (const float*)d_in[8];
  const float* bo  = (const float*)d_in[9];
  const float* Wa  = (const float*)d_in[10];
  const float* ba  = (const float*)d_in[11];
  float* out = (float*)d_out;

  char* ws = (char*)d_ws;
  const size_t TOK = (size_t)BATCH * SEQ;            // 8192
  _Float16* xh  = (_Float16*)(ws);                   // 4 MB
  _Float16* Wtq = (_Float16*)(ws + TOK * EMBED * 2);
  _Float16* Wtk = Wtq + EMBED * EMBED;
  _Float16* Wtv = Wtk + EMBED * EMBED;
  _Float16* Wto = Wtv + EMBED * EMBED;
  _Float16* Q   = Wto + EMBED * EMBED;               // [B][H][N][D]
  _Float16* K   = Q + TOK * EMBED;                   // (TOK*EMBED == B*H*N*D)
  _Float16* Vt  = K + TOK * EMBED;                   // [B][H][D][N]
  _Float16* ctx = Vt + TOK * EMBED;                  // [B*N][E]

  int nx = (int)(TOK * EMBED);
  gt_f32_to_f16<<<(nx + 255) / 256, 256, 0, stream>>>(x, xh, nx);
  gt_transpose_w<<<256, 256, 0, stream>>>(Wq, Wtq);
  gt_transpose_w<<<256, 256, 0, stream>>>(Wk, Wtk);
  gt_transpose_w<<<256, 256, 0, stream>>>(Wv, Wtv);
  gt_transpose_w<<<256, 256, 0, stream>>>(Wo, Wto);

  gt_qkv<<<dim3(EMBED / 16, TOK / 16, 3), 32, 0, stream>>>(
      xh, Wtq, Wtk, Wtv, bq, bk, bv, Q, K, Vt);

  gt_attn<<<dim3(BATCH * (SEQ / 16)), 256, 0, stream>>>(Q, K, Vt, adj, Wa, ba, ctx);

  gt_out<<<dim3(EMBED / 16, TOK / 16), 32, 0, stream>>>(ctx, Wto, bo, x, out);
}
